// CGCNN_2448131359387
// MI455X (gfx1250) — matrix operand order repaired and checked by reading, not compiled, not run on hardware
//
#include <hip/hip_runtime.h>
#include <hip/hip_fp16.h>

typedef __attribute__((ext_vector_type(16))) _Float16 v16h;
typedef __attribute__((ext_vector_type(8)))  _Float16 v8h;
typedef __attribute__((ext_vector_type(8)))  float    v8f;

#define NN   50000
#define NEg  100000     // line-graph node count (original edges)
#define LGE  200000     // line-graph edges
#define MPAD 100096     // NEg padded to multiple of 128 (GEMM M tile is 64)
#define EPS_BN 1e-6f
#define EPS_LN 1e-5f

__device__ __forceinline__ float lrelu01(float x){ return x > 0.f ? x : 0.01f*x; }
__device__ __forceinline__ float lrelu02(float x){ return x > 0.f ? x : 0.2f*x; }
__device__ __forceinline__ unsigned flipf(float f){
  unsigned u = __float_as_uint(f);
  return (u & 0x80000000u) ? ~u : (u | 0x80000000u);
}
__device__ __forceinline__ float unflipf(unsigned u){
  return __uint_as_float((u & 0x80000000u) ? (u & 0x7fffffffu) : ~u);
}

// ---------------------------------------------------------------- utilities
__global__ void zero_f32_k(float* p, size_t n){
  size_t i = (size_t)blockIdx.x*256 + threadIdx.x; if (i < n) p[i] = 0.f;
}
__global__ void fill_u32_k(unsigned* p, unsigned v, size_t n){
  size_t i = (size_t)blockIdx.x*256 + threadIdx.x; if (i < n) p[i] = v;
}
__global__ void cvt16_k(const float* __restrict__ X, _Float16* __restrict__ Y, size_t n){
  size_t i = (size_t)blockIdx.x*256 + threadIdx.x; if (i < n) Y[i] = (_Float16)X[i];
}

// column stats: sum & sumsq per column (population stats for BN)
__global__ void colstats_k(const float* __restrict__ X, int rows, int cols, int ld,
                           float* __restrict__ sum, float* __restrict__ sq){
  __shared__ float ss[8][32], ss2[8][32];
  int tx = threadIdx.x & 31, ty = threadIdx.x >> 5;
  int c = blockIdx.x*32 + tx;
  float s = 0.f, s2 = 0.f;
  if (c < cols){
    for (int r = blockIdx.y*8 + ty; r < rows; r += gridDim.y*8){
      float v = X[(size_t)r*ld + c]; s += v; s2 += v*v;
    }
  }
  ss[ty][tx] = s; ss2[ty][tx] = s2;
  __syncthreads();
  if (ty == 0 && c < cols){
    float a = 0.f, a2 = 0.f;
    #pragma unroll
    for (int i = 0; i < 8; ++i){ a += ss[i][tx]; a2 += ss2[i][tx]; }
    atomicAdd(&sum[c], a); atomicAdd(&sq[c], a2);
  }
}
__global__ void bn_affine_k(const float* sum, const float* sq, const float* g, const float* b,
                            float* A, float* B, int C, float invN){
  int c = blockIdx.x*256 + threadIdx.x; if (c >= C) return;
  float mu = sum[c]*invN, var = sq[c]*invN - mu*mu;
  float a = g[c]*rsqrtf(var + EPS_BN);
  A[c] = a; B[c] = b[c] - mu*a;
}
__global__ void bn_apply_k(float* X, const float* A, const float* B, size_t n, int N){
  size_t i = (size_t)blockIdx.x*256 + threadIdx.x; if (i >= n) return;
  int c = (int)(i % N); X[i] = X[i]*A[c] + B[c];
}
__global__ void lrelu_k(float* X, size_t n){
  size_t i = (size_t)blockIdx.x*256 + threadIdx.x; if (i < n) X[i] = lrelu01(X[i]);
}
__global__ void res_lrelu_k(float* X, const float* G, size_t n){
  size_t i = (size_t)blockIdx.x*256 + threadIdx.x; if (i < n) X[i] = lrelu01(X[i] + G[i]);
}
__global__ void add_bias_k(float* X, const float* b, size_t n, int N){
  size_t i = (size_t)blockIdx.x*256 + threadIdx.x; if (i >= n) return;
  X[i] += b[(int)(i % N)];
}

// row LayerNorm, cols in {256,1024}, block=256
__global__ void layernorm_k(float* X, const float* g, const float* b, int cols){
  __shared__ float red[8], red2[8], sh_mu, sh_ri;
  int row = blockIdx.x, t = threadIdx.x;
  int per = cols >> 8;
  size_t base = (size_t)row*cols;
  float vals[4]; float s = 0.f, s2 = 0.f;
  for (int i = 0; i < per; ++i){ float v = X[base + t + (i<<8)]; vals[i] = v; s += v; s2 += v*v; }
  for (int o = 16; o; o >>= 1){ s += __shfl_down(s, o, 32); s2 += __shfl_down(s2, o, 32); }
  if ((t & 31) == 0){ red[t>>5] = s; red2[t>>5] = s2; }
  __syncthreads();
  if (t == 0){
    float a = 0.f, a2 = 0.f;
    #pragma unroll
    for (int i = 0; i < 8; ++i){ a += red[i]; a2 += red2[i]; }
    float mu = a/cols, var = a2/cols - mu*mu;
    sh_mu = mu; sh_ri = rsqrtf(var + EPS_LN);
  }
  __syncthreads();
  for (int i = 0; i < per; ++i){
    int c = t + (i<<8);
    X[base + c] = (vals[i] - sh_mu)*sh_ri*g[c] + b[c];
  }
}

// ---------------------------------------------------------------- front end
__global__ void gather_emb_k(const int* xi, const float* emb, float* out){
  size_t i = (size_t)blockIdx.x*256 + threadIdx.x;
  if (i >= (size_t)NN*200) return;
  int r = (int)(i/200), c = (int)(i%200);
  out[i] = emb[(size_t)xi[r]*200 + c];
}
__global__ void ef_pre_k(const int* ei, const float* pos, const float* W, const float* bias, float* out){
  size_t i = (size_t)blockIdx.x*256 + threadIdx.x;
  if (i >= (size_t)NEg*200) return;
  int e = (int)(i/200), c = (int)(i%200);
  int r = ei[e]-1, cc = ei[NEg+e]-1;
  float d0 = pos[(size_t)cc*3+0]-pos[(size_t)r*3+0];
  float d1 = pos[(size_t)cc*3+1]-pos[(size_t)r*3+1];
  float d2 = pos[(size_t)cc*3+2]-pos[(size_t)r*3+2];
  out[i] = d0*W[c*3+0] + d1*W[c*3+1] + d2*W[c*3+2] + bias[c];
}
__global__ void build_h0_k(const int* ei, const float* hn_pre, const float* ef_pre,
                           const float* a5, const float* b5, const float* a4, const float* b4,
                           _Float16* h0){
  size_t i = (size_t)blockIdx.x*256 + threadIdx.x;
  if (i >= (size_t)NEg*608) return;
  int e = (int)(i/608), c = (int)(i%608);
  float v = 0.f;
  if (c < 200){ int r = ei[e]-1;      v = hn_pre[(size_t)r*200 + c]*a5[c] + b5[c]; }
  else if (c < 400){ int cc = ei[NEg+e]-1; int c2 = c-200; v = hn_pre[(size_t)cc*200 + c2]*a5[c2] + b5[c2]; }
  else if (c < 600){ int c2 = c-400;  v = ef_pre[(size_t)e*200 + c2]*a4[c2] + b4[c2]; }
  h0[i] = (_Float16)v;
}
__global__ void deg_count_k(const int* y1, int* deg){
  size_t i = (size_t)blockIdx.x*256 + threadIdx.x;
  if (i < (size_t)LGE) atomicAdd(&deg[y1[i]], 1);
}
__global__ void dinv_k(const int* deg, float* dinv){
  size_t i = (size_t)blockIdx.x*256 + threadIdx.x;
  if (i < (size_t)NEg) dinv[i] = rsqrtf((float)deg[i]);
}
__global__ void wconv_k(const float* W, _Float16* Wh, int K, int Nn, int Kpad, int Npad, int trans){
  size_t i = (size_t)blockIdx.x*256 + threadIdx.x;
  if (i >= (size_t)Kpad*Npad) return;
  int k = (int)(i/Npad), n = (int)(i%Npad);
  float v = 0.f;
  if (k < K && n < Nn) v = trans ? W[(size_t)n*K + k] : W[(size_t)k*Nn + n];
  Wh[i] = (_Float16)v;
}

// ---------------------------------------------------------------- WMMA GEMM
// C[Mpad,Npad] = A[Mpad,Kpad](f16,row-major) x B[Kpad,Npad](f16,row-major), fp32 out.
// 8 waves/block as 2(M)x4(N); wave computes a 32x64 strip: 2 A-fragments x 4 B-fragments,
// each B fragment feeds 2 WMMAs. Block tile 64x256 -> N=256 GEMMs read A exactly once.
__global__ __launch_bounds__(256) void wmma_gemm_k(const _Float16* __restrict__ A,
                                                   const _Float16* __restrict__ B,
                                                   float* __restrict__ C,
                                                   int lda, int ldb, int ldc, int Kpad){
  const int wave = threadIdx.x >> 5;
  const int lane = threadIdx.x & 31;
  const int wm = wave >> 2, wn = wave & 3;
  const int m0 = blockIdx.x*64 + wm*32;
  const int n0 = blockIdx.y*256 + wn*64;
  const int mrow  = lane & 15;
  const int kbase = (lane >> 4) << 3;            // 0 or 8 (A-frag K offset per ISA layout)
  v8f acc[2][4];
  #pragma unroll
  for (int mt = 0; mt < 2; ++mt)
    #pragma unroll
    for (int nt = 0; nt < 4; ++nt) acc[mt][nt] = (v8f){};
  const _Float16* arow0 = A + (size_t)(m0 + mrow)*lda + kbase;
  const _Float16* arow1 = arow0 + (size_t)16*lda;
  for (int k0 = 0; k0 < Kpad; k0 += 32){
    __builtin_prefetch(arow0 + k0 + 128, 0, 1);
    __builtin_prefetch(arow1 + k0 + 128, 0, 1);
    v8h lo0 = *(const v8h*)(arow0 + k0);
    v8h hi0 = *(const v8h*)(arow0 + k0 + 16);
    v8h lo1 = *(const v8h*)(arow1 + k0);
    v8h hi1 = *(const v8h*)(arow1 + k0 + 16);
    v16h a0, a1;
    #pragma unroll
    for (int j = 0; j < 8; ++j){ a0[j] = lo0[j]; a0[j+8] = hi0[j]; a1[j] = lo1[j]; a1[j+8] = hi1[j]; }
    const _Float16* brow = B + (size_t)(k0 + kbase)*ldb + n0 + (lane & 15);   // lane = N col, K in table order
    __builtin_prefetch(brow + (size_t)32*ldb, 0, 1);
    #pragma unroll
    for (int nt = 0; nt < 4; ++nt){
      v16h bf;
      #pragma unroll
      for (int j = 0; j < 8; ++j){ bf[j] = brow[(size_t)j*ldb + 16*nt]; bf[j+8] = brow[(size_t)(16 + j)*ldb + 16*nt]; }
      acc[0][nt] = __builtin_amdgcn_wmma_f32_16x16x32_f16(false, a0, false, bf, (short)0, acc[0][nt], false, false);
      acc[1][nt] = __builtin_amdgcn_wmma_f32_16x16x32_f16(false, a1, false, bf, (short)0, acc[1][nt], false, false);
    }
  }
  const int ccol = n0 + (lane & 15);
  #pragma unroll
  for (int mt = 0; mt < 2; ++mt){
    const int crow = m0 + mt*16 + ((lane >> 4) << 3);
    #pragma unroll
    for (int r = 0; r < 8; ++r){
      float* cp = C + (size_t)(crow + r)*ldc + ccol;
      #pragma unroll
      for (int nt = 0; nt < 4; ++nt) cp[16*nt] = acc[mt][nt][r];
    }
  }
}

// ---------------------------------------------------------------- GCN aggregate
__global__ void gcn_self_k(const float* XW, const float* bias, const float* dinv,
                           float* OUT, int N){
  size_t i = (size_t)blockIdx.x*256 + threadIdx.x;
  if (i >= (size_t)NEg*N) return;
  int r = (int)(i/N), c = (int)(i%N);
  float di = dinv[r];
  OUT[i] = bias[c] + XW[i]*di*di;
}
__global__ void gcn_edge_k(const int* y, const float* XW, const float* dinv,
                           float* OUT, int N){
  size_t i = (size_t)blockIdx.x*256 + threadIdx.x;
  if (i >= (size_t)LGE*N) return;
  int e = (int)(i/N), c = (int)(i%N);
  int s = y[e], d = y[LGE+e];
  atomicAdd(&OUT[(size_t)d*N + c], XW[(size_t)s*N + c]*dinv[s]*dinv[d]);
}

// ---------------------------------------------------------------- GAT
__global__ void gat_sd_k(const float* XW, const float* asrc, const float* adst,
                         float* S, float* D){
  int row = blockIdx.x;
  int h = threadIdx.x >> 5, lane = threadIdx.x & 31;
  size_t base = (size_t)row*1024 + h*256;
  float s = 0.f, d = 0.f;
  for (int c = lane; c < 256; c += 32){
    float v = XW[base + c];
    s += v*asrc[h*256 + c]; d += v*adst[h*256 + c];
  }
  for (int o = 16; o; o >>= 1){ s += __shfl_down(s, o, 32); d += __shfl_down(d, o, 32); }
  if (lane == 0){ S[row*4 + h] = s; D[row*4 + h] = d; }
}
__global__ void gat_emax_edge_k(const int* y, const float* S, const float* D, unsigned* EMAX){
  size_t i = (size_t)blockIdx.x*256 + threadIdx.x;
  if (i >= (size_t)LGE*4) return;
  int e = (int)(i >> 2), h = (int)(i & 3);
  int s = y[e], d = y[LGE+e];
  atomicMax(&EMAX[d*4 + h], flipf(lrelu02(S[s*4+h] + D[d*4+h])));
}
__global__ void gat_emax_self_k(const float* S, const float* D, unsigned* EMAX){
  size_t i = (size_t)blockIdx.x*256 + threadIdx.x;
  if (i >= (size_t)NEg*4) return;
  atomicMax(&EMAX[i], flipf(lrelu02(S[i] + D[i])));
}
__global__ void gat_ee_edge_k(const int* y, const float* S, const float* D,
                              const unsigned* EMAX, float* ALPHA, float* DEN){
  size_t i = (size_t)blockIdx.x*256 + threadIdx.x;
  if (i >= (size_t)LGE*4) return;
  int e = (int)(i >> 2), h = (int)(i & 3);
  int s = y[e], d = y[LGE+e];
  float v = lrelu02(S[s*4+h] + D[d*4+h]);
  float ee = __expf(v - unflipf(EMAX[d*4 + h]));
  ALPHA[i] = ee;
  atomicAdd(&DEN[d*4 + h], ee);
}
__global__ void gat_ee_self_k(const float* S, const float* D, const unsigned* EMAX,
                              float* ALPHA, float* DEN){
  size_t i = (size_t)blockIdx.x*256 + threadIdx.x;
  if (i >= (size_t)NEg*4) return;
  float v = lrelu02(S[i] + D[i]);
  float ee = __expf(v - unflipf(EMAX[i]));
  ALPHA[(size_t)LGE*4 + i] = ee;
  atomicAdd(&DEN[i], ee);
}
__global__ void gat_agg_edge_k(const int* y, const float* XW, const float* ALPHA,
                               const float* DEN, float* OUT){
  size_t i = (size_t)blockIdx.x*256 + threadIdx.x;
  if (i >= (size_t)LGE*1024) return;
  int e = (int)(i >> 10), j = (int)(i & 1023), h = j >> 8;
  int s = y[e], d = y[LGE+e];
  float alpha = ALPHA[(size_t)e*4 + h] / DEN[d*4 + h];
  atomicAdd(&OUT[(size_t)d*1024 + j], XW[(size_t)s*1024 + j]*alpha);
}
__global__ void gat_agg_self_k(const float* XW, const float* ALPHA, const float* DEN, float* OUT){
  size_t i = (size_t)blockIdx.x*256 + threadIdx.x;
  if (i >= (size_t)NEg*1024) return;
  int r = (int)(i >> 10), j = (int)(i & 1023), h = j >> 8;
  float alpha = ALPHA[(size_t)LGE*4 + r*4 + h] / DEN[r*4 + h];
  atomicAdd(&OUT[i], XW[i]*alpha);
}

// ---------------------------------------------------------------- heads
__global__ void fc1_k(const float* H, const float* W, const float* b, float* out){
  int row = blockIdx.x*8 + (threadIdx.x >> 5);
  int lane = threadIdx.x & 31;
  if (row >= NEg) return;
  float s = 0.f;
  for (int c = lane; c < 256; c += 32) s += H[(size_t)row*256 + c]*W[c];
  for (int o = 16; o; o >>= 1) s += __shfl_down(s, o, 32);
  if (lane == 0) out[row] = lrelu01(s + b[0]);
}
__global__ void fc2_fin_k(const float* XW, const float* b, float* out){
  size_t i = (size_t)blockIdx.x*256 + threadIdx.x;
  if (i >= (size_t)NEg*200) return;
  int r = (int)(i/200), c = (int)(i%200);
  out[(size_t)NEg + i] = lrelu01(XW[(size_t)r*256 + c] + b[c]);
}

// ---------------------------------------------------------------- host
static inline unsigned cdivu(size_t a, size_t b){ return (unsigned)((a + b - 1)/b); }
static inline char* wsoff(void* ws, size_t& off, size_t bytes){
  char* p = (char*)ws + off;
  off += (bytes + 255) & ~(size_t)255;
  return p;
}

extern "C" void kernel_launch(void* const* d_in, const int* in_sizes, int n_in,
                              void* d_out, int out_size, void* d_ws, size_t ws_size,
                              hipStream_t stream){
  (void)in_sizes; (void)n_in; (void)out_size; (void)ws_size;
  const int*   x_idx  = (const int*)  d_in[0];
  const int*   ei     = (const int*)  d_in[1];
  const float* pos    = (const float*)d_in[2];
  const int*   y      = (const int*)  d_in[3];
  const float* emb    = (const float*)d_in[4];
  const float* fc3_W  = (const float*)d_in[5];
  const float* fc3_b  = (const float*)d_in[6];
  const float* bn4_g  = (const float*)d_in[7];
  const float* bn4_b  = (const float*)d_in[8];
  const float* bn5_g  = (const float*)d_in[9];
  const float* bn5_b  = (const float*)d_in[10];
  const float* conv1_W= (const float*)d_in[11];
  const float* conv1_b= (const float*)d_in[12];
  const float* bn1_g  = (const float*)d_in[13];
  const float* bn1_b  = (const float*)d_in[14];
  const float* ln1_g  = (const float*)d_in[15];
  const float* ln1_b  = (const float*)d_in[16];
  const float* conv2_W= (const float*)d_in[17];
  const float* conv2_b= (const float*)d_in[18];
  const float* bn2_g  = (const float*)d_in[19];
  const float* bn2_b  = (const float*)d_in[20];
  const float* gat_W  = (const float*)d_in[21];
  const float* gat_as = (const float*)d_in[22];
  const float* gat_ad = (const float*)d_in[23];
  const float* gat_bias=(const float*)d_in[24];
  const float* ln2_g  = (const float*)d_in[25];
  const float* ln2_b  = (const float*)d_in[26];
  const float* conv3_W= (const float*)d_in[27];
  const float* conv3_b= (const float*)d_in[28];
  const float* bn3_g  = (const float*)d_in[29];
  const float* bn3_b  = (const float*)d_in[30];
  const float* fc1_W  = (const float*)d_in[31];
  const float* fc1_b  = (const float*)d_in[32];
  const float* fc2_W  = (const float*)d_in[33];
  const float* fc2_b  = (const float*)d_in[34];
  float* out = (float*)d_out;

  size_t off = 0;
  float*    st_sum = (float*)   wsoff(d_ws, off, 1024*4);
  float*    st_sq  = (float*)   wsoff(d_ws, off, 1024*4);
  float*    a5     = (float*)   wsoff(d_ws, off, 256*4);
  float*    b5     = (float*)   wsoff(d_ws, off, 256*4);
  float*    a4     = (float*)   wsoff(d_ws, off, 256*4);
  float*    b4     = (float*)   wsoff(d_ws, off, 256*4);
  float*    bnA    = (float*)   wsoff(d_ws, off, 256*4);
  float*    bnB    = (float*)   wsoff(d_ws, off, 256*4);
  int*      deg    = (int*)     wsoff(d_ws, off, (size_t)MPAD*4);
  float*    dinv   = (float*)   wsoff(d_ws, off, (size_t)MPAD*4);
  float*    Sbuf   = (float*)   wsoff(d_ws, off, (size_t)MPAD*4*4);
  float*    Dbuf   = (float*)   wsoff(d_ws, off, (size_t)MPAD*4*4);
  unsigned* EMAX   = (unsigned*)wsoff(d_ws, off, (size_t)MPAD*4*4);
  float*    DEN    = (float*)   wsoff(d_ws, off, (size_t)MPAD*4*4);
  float*    ALPHA  = (float*)   wsoff(d_ws, off, (size_t)(LGE+NEg)*4*4);
  float*    hn_pre = (float*)   wsoff(d_ws, off, (size_t)NN*200*4);
  float*    ef_pre = (float*)   wsoff(d_ws, off, (size_t)NEg*200*4);
  _Float16* h0     = (_Float16*)wsoff(d_ws, off, (size_t)MPAD*608*2);
  _Float16* Wh     = (_Float16*)wsoff(d_ws, off, (size_t)1024*1024*2);
  _Float16* Hf16   = (_Float16*)wsoff(d_ws, off, (size_t)MPAD*1024*2);
  float*    XW     = (float*)   wsoff(d_ws, off, (size_t)MPAD*1024*4);
  float*    AG     = (float*)   wsoff(d_ws, off, (size_t)MPAD*1024*4);
  float*    Hb     = (float*)   wsoff(d_ws, off, (size_t)MPAD*1024*4);

  const dim3 B256(256);
  #define ELN(n) dim3(cdivu((n),256))

  // --- degrees / symmetric norm of line graph (with self-loops) ---
  fill_u32_k<<<ELN((size_t)MPAD), B256, 0, stream>>>((unsigned*)deg, 1u, (size_t)MPAD);
  deg_count_k<<<ELN((size_t)LGE), B256, 0, stream>>>(y + LGE, deg);
  dinv_k<<<ELN((size_t)NEg), B256, 0, stream>>>(deg, dinv);

  // --- node embedding BN (bn5) ---
  gather_emb_k<<<ELN((size_t)NN*200), B256, 0, stream>>>(x_idx, emb, hn_pre);
  zero_f32_k<<<ELN((size_t)2048), B256, 0, stream>>>(st_sum, 2048);   // sum+sq contiguous
  colstats_k<<<dim3(cdivu(200,32),64), B256, 0, stream>>>(hn_pre, NN, 200, 200, st_sum, st_sq);
  bn_affine_k<<<dim3(1), B256, 0, stream>>>(st_sum, st_sq, bn5_g, bn5_b, a5, b5, 200, 1.0f/NN);

  // --- edge geometric features + BN (bn4) ---
  ef_pre_k<<<ELN((size_t)NEg*200), B256, 0, stream>>>(ei, pos, fc3_W, fc3_b, ef_pre);
  zero_f32_k<<<ELN((size_t)2048), B256, 0, stream>>>(st_sum, 2048);
  colstats_k<<<dim3(cdivu(200,32),64), B256, 0, stream>>>(ef_pre, NEg, 200, 200, st_sum, st_sq);
  bn_affine_k<<<dim3(1), B256, 0, stream>>>(st_sum, st_sq, bn4_g, bn4_b, a4, b4, 200, 1.0f/NEg);

  // --- h0 = concat(hn[row], hn[col], ef) in f16, K padded 600->608 ---
  build_h0_k<<<ELN((size_t)NEg*608), B256, 0, stream>>>(ei, hn_pre, ef_pre, a5, b5, a4, b4, h0);

  // ================= conv1 (GCN 608->256) =================
  wconv_k<<<ELN((size_t)608*256), B256, 0, stream>>>(conv1_W, Wh, 600, 256, 608, 256, 0);
  wmma_gemm_k<<<dim3(MPAD/64, 1), B256, 0, stream>>>(h0, Wh, XW, 608, 256, 256, 608);
  gcn_self_k<<<ELN((size_t)NEg*256), B256, 0, stream>>>(XW, conv1_b, dinv, AG, 256);
  gcn_edge_k<<<ELN((size_t)LGE*256), B256, 0, stream>>>(y, XW, dinv, AG, 256);
  lrelu_k<<<ELN((size_t)NEg*256), B256, 0, stream>>>(AG, (size_t)NEg*256);
  zero_f32_k<<<ELN((size_t)2048), B256, 0, stream>>>(st_sum, 2048);
  colstats_k<<<dim3(8,64), B256, 0, stream>>>(AG, NEg, 256, 256, st_sum, st_sq);
  bn_affine_k<<<dim3(1), B256, 0, stream>>>(st_sum, st_sq, bn1_g, bn1_b, bnA, bnB, 256, 1.0f/NEg);
  bn_apply_k<<<ELN((size_t)NEg*256), B256, 0, stream>>>(AG, bnA, bnB, (size_t)NEg*256, 256);
  layernorm_k<<<dim3(NEg), B256, 0, stream>>>(AG, ln1_g, ln1_b, 256);

  // ================= conv2 (GCN 256->256, residual) =================
  cvt16_k<<<ELN((size_t)NEg*256), B256, 0, stream>>>(AG, Hf16, (size_t)NEg*256);
  wconv_k<<<ELN((size_t)256*256), B256, 0, stream>>>(conv2_W, Wh, 256, 256, 256, 256, 0);
  wmma_gemm_k<<<dim3(MPAD/64, 1), B256, 0, stream>>>(Hf16, Wh, XW, 256, 256, 256, 256);
  gcn_self_k<<<ELN((size_t)NEg*256), B256, 0, stream>>>(XW, conv2_b, dinv, Hb, 256);
  gcn_edge_k<<<ELN((size_t)LGE*256), B256, 0, stream>>>(y, XW, dinv, Hb, 256);
  res_lrelu_k<<<ELN((size_t)NEg*256), B256, 0, stream>>>(AG, Hb, (size_t)NEg*256);
  zero_f32_k<<<ELN((size_t)2048), B256, 0, stream>>>(st_sum, 2048);
  colstats_k<<<dim3(8,64), B256, 0, stream>>>(AG, NEg, 256, 256, st_sum, st_sq);
  bn_affine_k<<<dim3(1), B256, 0, stream>>>(st_sum, st_sq, bn2_g, bn2_b, bnA, bnB, 256, 1.0f/NEg);
  bn_apply_k<<<ELN((size_t)NEg*256), B256, 0, stream>>>(AG, bnA, bnB, (size_t)NEg*256, 256);

  // ================= GAT (256 -> 4x256) =================
  cvt16_k<<<ELN((size_t)NEg*256), B256, 0, stream>>>(AG, Hf16, (size_t)NEg*256);
  wconv_k<<<ELN((size_t)256*1024), B256, 0, stream>>>(gat_W, Wh, 256, 1024, 256, 1024, 0);
  wmma_gemm_k<<<dim3(MPAD/64, 4), B256, 0, stream>>>(Hf16, Wh, XW, 256, 1024, 1024, 256);
  gat_sd_k<<<dim3(NEg), dim3(128), 0, stream>>>(XW, gat_as, gat_ad, Sbuf, Dbuf);
  {
    float neg = -3.0e38f; unsigned nu = ~(*(unsigned*)&neg);   // flip of large negative
    fill_u32_k<<<ELN((size_t)NEg*4), B256, 0, stream>>>(EMAX, nu, (size_t)NEg*4);
  }
  gat_emax_edge_k<<<ELN((size_t)LGE*4), B256, 0, stream>>>(y, Sbuf, Dbuf, EMAX);
  gat_emax_self_k<<<ELN((size_t)NEg*4), B256, 0, stream>>>(Sbuf, Dbuf, EMAX);
  zero_f32_k<<<ELN((size_t)NEg*4), B256, 0, stream>>>(DEN, (size_t)NEg*4);
  gat_ee_edge_k<<<ELN((size_t)LGE*4), B256, 0, stream>>>(y, Sbuf, Dbuf, EMAX, ALPHA, DEN);
  gat_ee_self_k<<<ELN((size_t)NEg*4), B256, 0, stream>>>(Sbuf, Dbuf, EMAX, ALPHA, DEN);
  zero_f32_k<<<ELN((size_t)NEg*1024), B256, 0, stream>>>(Hb, (size_t)NEg*1024);
  gat_agg_edge_k<<<ELN((size_t)LGE*1024), B256, 0, stream>>>(y, XW, ALPHA, DEN, Hb);
  gat_agg_self_k<<<ELN((size_t)NEg*1024), B256, 0, stream>>>(XW, ALPHA, DEN, Hb);
  add_bias_k<<<ELN((size_t)NEg*1024), B256, 0, stream>>>(Hb, gat_bias, (size_t)NEg*1024, 1024);
  layernorm_k<<<dim3(NEg), B256, 0, stream>>>(Hb, ln2_g, ln2_b, 1024);

  // ================= conv3 (GCN 1024->256) =================
  cvt16_k<<<ELN((size_t)NEg*1024), B256, 0, stream>>>(Hb, Hf16, (size_t)NEg*1024);
  wconv_k<<<ELN((size_t)1024*256), B256, 0, stream>>>(conv3_W, Wh, 1024, 256, 1024, 256, 0);
  wmma_gemm_k<<<dim3(MPAD/64, 1), B256, 0, stream>>>(Hf16, Wh, XW, 1024, 256, 256, 1024);
  gcn_self_k<<<ELN((size_t)NEg*256), B256, 0, stream>>>(XW, conv3_b, dinv, AG, 256);
  gcn_edge_k<<<ELN((size_t)LGE*256), B256, 0, stream>>>(y, XW, dinv, AG, 256);
  lrelu_k<<<ELN((size_t)NEg*256), B256, 0, stream>>>(AG, (size_t)NEg*256);
  zero_f32_k<<<ELN((size_t)2048), B256, 0, stream>>>(st_sum, 2048);
  colstats_k<<<dim3(8,64), B256, 0, stream>>>(AG, NEg, 256, 256, st_sum, st_sq);
  bn_affine_k<<<dim3(1), B256, 0, stream>>>(st_sum, st_sq, bn3_g, bn3_b, bnA, bnB, 256, 1.0f/NEg);
  bn_apply_k<<<ELN((size_t)NEg*256), B256, 0, stream>>>(AG, bnA, bnB, (size_t)NEg*256, 256);

  // ================= heads =================
  fc1_k<<<dim3(cdivu(NEg,8)), B256, 0, stream>>>(AG, fc1_W, fc1_b, out);
  cvt16_k<<<ELN((size_t)NEg*256), B256, 0, stream>>>(AG, Hf16, (size_t)NEg*256);
  wconv_k<<<ELN((size_t)256*256), B256, 0, stream>>>(fc2_W, Wh, 256, 200, 256, 256, 1);
  wmma_gemm_k<<<dim3(MPAD/64, 1), B256, 0, stream>>>(Hf16, Wh, XW, 256, 256, 256, 256);
  fc2_fin_k<<<ELN((size_t)NEg*200), B256, 0, stream>>>(XW, fc2_b, out);
  #undef ELN
}
